// PairUpdate_63788854280316
// MI455X (gfx1250) — compile-verified, matrix operand order checked by reading, not run on hardware
//
#include <hip/hip_runtime.h>
#include <hip/hip_bf16.h>

// ---------------------------------------------------------------------------
// PairUpdate (triangle attention over columns) for MI455X / gfx1250.
// All GEMMs run on v_wmma_f32_16x16x32_bf16 (wave32 WMMA) with A-operand
// reuse across 4-8 N-tiles per wave. fp32 only for LN / softmax / residuals.
// ---------------------------------------------------------------------------

#define Lsz 256
#define Csz 256
#define Hsz 8
#define Dsz 32
#define Fsz 512
#define Msz (Lsz * Lsz)           // 65536 rows
#define SCALE_QK 0.17677669529663687f   // 1/sqrt(32)
#define NEG_BIG (-3.0e38f)

typedef __attribute__((ext_vector_type(16))) __bf16 v16bf;
typedef __attribute__((ext_vector_type(8)))  __bf16 v8bf;
typedef __attribute__((ext_vector_type(8)))  float  v8f;

__device__ __forceinline__ __bf16 f2bf(float f) {
  unsigned u = __builtin_bit_cast(unsigned, f);
  u += 0x7FFFu + ((u >> 16) & 1u);          // round-to-nearest-even
  unsigned short h = (unsigned short)(u >> 16);
  return __builtin_bit_cast(__bf16, h);
}

// Load one WMMA A/B operand tile (16 rows x 32 K) from a row-major bf16
// matrix. Per ISA 16-bit operand layout: lane L holds row (L&15); vector
// elements 0..7 = K chunk at k0 + 8*(L>=16), elements 8..15 = that + 16.
// Both chunks are contiguous 16-byte loads.
__device__ __forceinline__ v16bf ld_tile(const __bf16* base, int ld,
                                         int rowbase, int kbase, int lane) {
  const int r  = rowbase + (lane & 15);
  const int k0 = kbase + ((lane >> 4) << 3);
  const __bf16* p = base + (size_t)r * ld + k0;
  v8bf lo = *(const v8bf*)p;
  v8bf hi = *(const v8bf*)(p + 16);
  v16bf out;
#pragma unroll
  for (int i = 0; i < 8; ++i) { out[i] = lo[i]; out[i + 8] = hi[i]; }
  return out;
}

__device__ __forceinline__ v8f wmma_bf16(v16bf a, v16bf b, v8f c) {
  // (neg_a, A, neg_b, B, c_mod, C, reuse_a, reuse_b)
  return __builtin_amdgcn_wmma_f32_16x16x32_bf16(false, a, false, b,
                                                 (short)0, c, false, false);
}

// Multi-N-tile GEMM chain: one A load amortized over NT adjacent B tiles.
template <int NT, int KSTEPS>
__device__ __forceinline__ void gemm_tiles(const __bf16* __restrict__ A, int lda,
                                           const __bf16* __restrict__ B, int ldb,
                                           int nbase, int lane, v8f* acc) {
#pragma unroll
  for (int kk = 0; kk < KSTEPS; ++kk) {
    v16bf a = ld_tile(A, lda, 0, kk * 32, lane);
#pragma unroll
    for (int n = 0; n < NT; ++n) {
      v16bf b = ld_tile(B, ldb, nbase + n * 16, kk * 32, lane);
      acc[n] = wmma_bf16(a, b, acc[n]);
    }
  }
}

// ---------------------------------------------------------------------------
// Elementwise fp32 -> bf16 cast (weights).
// ---------------------------------------------------------------------------
__global__ void cast_f32_bf16_kernel(const float* __restrict__ src,
                                     __bf16* __restrict__ dst, int n) {
  int i = blockIdx.x * blockDim.x + threadIdx.x;
  if (i < n) dst[i] = f2bf(src[i]);
}

// ---------------------------------------------------------------------------
// LayerNorm over C=256 (one block per row) -> bf16.
// ---------------------------------------------------------------------------
__global__ void ln_cast_kernel(const float* __restrict__ x,
                               const float* __restrict__ g,
                               const float* __restrict__ be,
                               __bf16* __restrict__ out) {
  __shared__ float red[256];
  const int row = blockIdx.x;
  const int t   = threadIdx.x;
  float v = x[(size_t)row * Csz + t];
  red[t] = v;
  __syncthreads();
  for (int s = 128; s > 0; s >>= 1) {
    if (t < s) red[t] += red[t + s];
    __syncthreads();
  }
  const float mean = red[0] * (1.0f / Csz);
  __syncthreads();
  const float d = v - mean;
  red[t] = d * d;
  __syncthreads();
  for (int s = 128; s > 0; s >>= 1) {
    if (t < s) red[t] += red[t + s];
    __syncthreads();
  }
  const float rstd = rsqrtf(red[0] * (1.0f / Csz) + 1e-5f);
  out[(size_t)row * Csz + t] = f2bf(d * rstd * g[t] + be[t]);
}

// ---------------------------------------------------------------------------
// Q/K/V projection: y = xln @ W^T + b, written head-transposed as
// out[j][h][i][d] so attention reads contiguous [i x d] panels.
// grid = (M/16, 3), block = 128: 4 waves x (4 N-tiles = 64 cols) = 256 cols.
// ---------------------------------------------------------------------------
__global__ void qkv_proj_kernel(const __bf16* __restrict__ xln,
                                const __bf16* __restrict__ Wq,
                                const __bf16* __restrict__ Wk,
                                const __bf16* __restrict__ Wv,
                                const float* __restrict__ bq,
                                const float* __restrict__ bk,
                                const float* __restrict__ bv,
                                __bf16* __restrict__ qT,
                                __bf16* __restrict__ kT,
                                __bf16* __restrict__ vT) {
  __shared__ alignas(16) __bf16 atile[16 * Csz];
  const int tm    = blockIdx.x;
  const int which = blockIdx.y;
  const __bf16* W    = (which == 0) ? Wq : (which == 1) ? Wk : Wv;
  const float*  bias = (which == 0) ? bq : (which == 1) ? bk : bv;
  __bf16* outT       = (which == 0) ? qT : (which == 1) ? kT : vT;

  const int t = threadIdx.x, wave = t >> 5, lane = t & 31;

  // 16 consecutive rows -> contiguous 8 KB copy into LDS
  const __bf16* src = xln + (size_t)tm * 16 * Csz;
#pragma unroll
  for (int e = 0; e < 4; ++e)
    ((v8bf*)atile)[t * 4 + e] = ((const v8bf*)src)[t * 4 + e];
  __syncthreads();

  const int nbase = wave * 64;
  __builtin_prefetch(W + (size_t)nbase * Csz, 0, 3);
  v8f acc[4] = {};
  gemm_tiles<4, 8>(atile, Csz, W, Csz, nbase, lane, acc);

  const int mm0 = (lane >> 4) << 3;
  const int nn  = lane & 15;
#pragma unroll
  for (int nt = 0; nt < 4; ++nt)
#pragma unroll
    for (int r = 0; r < 8; ++r) {
      const int m = tm * 16 + r + mm0;
      const int n = nbase + nt * 16 + nn;
      const int i = m / Lsz, j = m % Lsz;
      const int h = n / Dsz, d = n % Dsz;
      outT[(size_t)j * (Hsz * Lsz * Dsz) + h * (Lsz * Dsz) + i * Dsz + d] =
          f2bf(acc[nt][r] + bias[n]);
    }
}

// ---------------------------------------------------------------------------
// Attention per (j, h): S = Q K^T * scale + bias, mask, softmax over k,
// O = P V. grid = (L, H), block = 128 (4 waves x 4 strips of 16 rows).
// ---------------------------------------------------------------------------
__global__ void attention_kernel(const __bf16* __restrict__ qT,
                                 const __bf16* __restrict__ kT,
                                 const __bf16* __restrict__ vT,
                                 const float* __restrict__ pair_bias,
                                 const int* __restrict__ mask,
                                 __bf16* __restrict__ attn) {
  __shared__ alignas(16) __bf16 vtile[Dsz * Lsz];        // V^T [d][k], 16 KB
  __shared__ alignas(16) __bf16 ptile[4][16 * Lsz];      // per-wave P, 8 KB each
  const int j = blockIdx.x, h = blockIdx.y;
  const int t = threadIdx.x, wave = t >> 5, lane = t & 31;

  const size_t base = (size_t)j * (Hsz * Lsz * Dsz) + (size_t)h * (Lsz * Dsz);
  const __bf16* Q = qT + base;   // [256 i][32 d]
  const __bf16* K = kT + base;
  const __bf16* V = vT + base;
  const float* bias = pair_bias + (size_t)h * Lsz * Lsz;

  // stage V^T: vtile[d][k] = V[k][d]
  for (int e = t; e < Dsz * Lsz; e += 128) {
    const int d = e >> 8, k = e & 255;
    vtile[d * Lsz + k] = V[k * Dsz + d];
  }
  __syncthreads();

  const int mm0 = (lane >> 4) << 3;
  const int nn  = lane & 15;
  __bf16* P = &ptile[wave][0];

  for (int strip = 0; strip < 4; ++strip) {
    const int ibase = (wave * 4 + strip) * 16;
    // A = Q strip (16 x 32), K-dim = d = 32 -> one WMMA per 16-col tile,
    // A operand reused across all 16 score tiles.
    v16bf aq = ld_tile(Q, Dsz, ibase, 0, lane);
    v8f s[16];
#pragma unroll
    for (int kt = 0; kt < 16; ++kt) {
      v16bf bk = ld_tile(K, Dsz, kt * 16, 0, lane);
      v8f z = {};
      s[kt] = wmma_bf16(aq, bk, z);
    }
    // scale + bias + mask, then rowwise softmax
    float rmax[8];
#pragma unroll
    for (int r = 0; r < 8; ++r) rmax[r] = NEG_BIG;
#pragma unroll
    for (int kt = 0; kt < 16; ++kt) {
#pragma unroll
      for (int r = 0; r < 8; ++r) {
        const int i = ibase + r + mm0;
        const int kc = kt * 16 + nn;
        float sc = s[kt][r] * SCALE_QK + bias[i * Lsz + kc];
        if (mask[i * Lsz + kc] == 0) sc = NEG_BIG;
        s[kt][r] = sc;
        rmax[r] = fmaxf(rmax[r], sc);
      }
    }
#pragma unroll
    for (int r = 0; r < 8; ++r) {      // rows live in 16-lane halves
      float m = rmax[r];
#pragma unroll
      for (int off = 8; off >= 1; off >>= 1) m = fmaxf(m, __shfl_xor(m, off, 16));
      rmax[r] = m;
    }
    float rsum[8];
#pragma unroll
    for (int r = 0; r < 8; ++r) rsum[r] = 0.0f;
#pragma unroll
    for (int kt = 0; kt < 16; ++kt)
#pragma unroll
      for (int r = 0; r < 8; ++r) {
        const float e = __expf(s[kt][r] - rmax[r]);
        s[kt][r] = e;
        rsum[r] += e;
      }
#pragma unroll
    for (int r = 0; r < 8; ++r) {
      float m = rsum[r];
#pragma unroll
      for (int off = 8; off >= 1; off >>= 1) m += __shfl_xor(m, off, 16);
      rsum[r] = 1.0f / m;
    }
    // P (bf16) -> wave-private LDS strip, row-major [16][256]
#pragma unroll
    for (int kt = 0; kt < 16; ++kt)
#pragma unroll
      for (int r = 0; r < 8; ++r)
        P[(r + mm0) * Lsz + kt * 16 + nn] = f2bf(s[kt][r] * rsum[r]);

    // O = P (16x256) @ V (256x32): both d-tiles share each A load of P.
    v8f acc[2] = {};
#pragma unroll
    for (int kk = 0; kk < 8; ++kk) {
      v16bf a = ld_tile(P, Lsz, 0, kk * 32, lane);
#pragma unroll
      for (int dt = 0; dt < 2; ++dt) {
        v16bf b = ld_tile(vtile, Lsz, dt * 16, kk * 32, lane);
        acc[dt] = wmma_bf16(a, b, acc[dt]);
      }
    }
#pragma unroll
    for (int dt = 0; dt < 2; ++dt)
#pragma unroll
      for (int r = 0; r < 8; ++r) {
        const int i = ibase + r + mm0;
        const int d = dt * 16 + nn;
        attn[((size_t)i * Lsz + j) * Csz + h * Dsz + d] = f2bf(acc[dt][r]);
      }
  }
}

// ---------------------------------------------------------------------------
// pr = pair_repr + attn @ Wo^T + bo   (writes fp32 into d_out)
// grid = M/16, block = 128: 4 waves x 4 N-tiles = 256 cols.
// ---------------------------------------------------------------------------
__global__ void outproj_kernel(const __bf16* __restrict__ attn,
                               const __bf16* __restrict__ Wo,
                               const float* __restrict__ bo,
                               const float* __restrict__ pair,
                               float* __restrict__ pr) {
  __shared__ alignas(16) __bf16 atile[16 * Csz];
  const int tm = blockIdx.x;
  const int t = threadIdx.x, wave = t >> 5, lane = t & 31;

  const __bf16* src = attn + (size_t)tm * 16 * Csz;
#pragma unroll
  for (int e = 0; e < 4; ++e)
    ((v8bf*)atile)[t * 4 + e] = ((const v8bf*)src)[t * 4 + e];
  __syncthreads();

  const int nbase = wave * 64;
  __builtin_prefetch(Wo + (size_t)nbase * Csz, 0, 3);
  v8f acc[4] = {};
  gemm_tiles<4, 8>(atile, Csz, Wo, Csz, nbase, lane, acc);

  const int mm0 = (lane >> 4) << 3;
  const int nn  = lane & 15;
#pragma unroll
  for (int nt = 0; nt < 4; ++nt)
#pragma unroll
    for (int r = 0; r < 8; ++r) {
      const int m = tm * 16 + r + mm0;
      const int n = nbase + nt * 16 + nn;
      pr[(size_t)m * Csz + n] = acc[nt][r] + bo[n] + pair[(size_t)m * Csz + n];
    }
}

// ---------------------------------------------------------------------------
// Fused FFN: out += relu(y @ W1^T + b1) @ W2^T + b2. Hidden tile (16x512)
// never leaves LDS. grid = M/16, block = 128.
// Phase 1: 4 waves x 8 N-tiles = 512 hidden cols. Phase 2: 4 waves x 4.
// ---------------------------------------------------------------------------
__global__ void ffn_kernel(const __bf16* __restrict__ y,
                           const __bf16* __restrict__ W1,
                           const __bf16* __restrict__ W2,
                           const float* __restrict__ b1,
                           const float* __restrict__ b2,
                           float* __restrict__ out) {
  __shared__ alignas(16) __bf16 ytile[16 * Csz];   // 8 KB
  __shared__ alignas(16) __bf16 htile[16 * Fsz];   // 16 KB
  const int tm = blockIdx.x;
  const int t = threadIdx.x, wave = t >> 5, lane = t & 31;

  const __bf16* src = y + (size_t)tm * 16 * Csz;
#pragma unroll
  for (int e = 0; e < 4; ++e)
    ((v8bf*)ytile)[t * 4 + e] = ((const v8bf*)src)[t * 4 + e];
  __syncthreads();

  const int mm0 = (lane >> 4) << 3;
  const int nn  = lane & 15;

  // hidden = relu(y @ W1^T + b1): wave covers 128 of 512 cols, A reused x8
  {
    const int nbase = wave * 128;
    __builtin_prefetch(W1 + (size_t)nbase * Csz, 0, 3);
    v8f acc[8] = {};
    gemm_tiles<8, 8>(ytile, Csz, W1, Csz, nbase, lane, acc);
#pragma unroll
    for (int nt = 0; nt < 8; ++nt)
#pragma unroll
      for (int r = 0; r < 8; ++r) {
        const int n = nbase + nt * 16 + nn;
        htile[(r + mm0) * Fsz + n] = f2bf(fmaxf(acc[nt][r] + b1[n], 0.0f));
      }
  }
  __syncthreads();

  // out += hidden @ W2^T + b2: wave covers 64 of 256 cols, K = 512 (16 steps)
  {
    const int nbase = wave * 64;
    __builtin_prefetch(W2 + (size_t)nbase * Fsz, 0, 3);
    v8f acc[4] = {};
    gemm_tiles<4, 16>(htile, Fsz, W2, Fsz, nbase, lane, acc);
#pragma unroll
    for (int nt = 0; nt < 4; ++nt)
#pragma unroll
      for (int r = 0; r < 8; ++r) {
        const int m = tm * 16 + r + mm0;
        const int n = nbase + nt * 16 + nn;
        out[(size_t)m * Csz + n] += acc[nt][r] + b2[n];
      }
  }
}

// ---------------------------------------------------------------------------
extern "C" void kernel_launch(void* const* d_in, const int* in_sizes, int n_in,
                              void* d_out, int out_size, void* d_ws, size_t ws_size,
                              hipStream_t stream) {
  (void)in_sizes; (void)n_in; (void)out_size; (void)ws_size;
  const float* pair_repr = (const float*)d_in[0];
  const float* pair_bias = (const float*)d_in[1];
  const float* Wq = (const float*)d_in[2];  const float* bq = (const float*)d_in[3];
  const float* Wk = (const float*)d_in[4];  const float* bk = (const float*)d_in[5];
  const float* Wv = (const float*)d_in[6];  const float* bv = (const float*)d_in[7];
  const float* Wo = (const float*)d_in[8];  const float* bo = (const float*)d_in[9];
  const float* W1 = (const float*)d_in[10]; const float* b1 = (const float*)d_in[11];
  const float* W2 = (const float*)d_in[12]; const float* b2 = (const float*)d_in[13];
  const float* g1 = (const float*)d_in[14]; const float* be1 = (const float*)d_in[15];
  const float* g2 = (const float*)d_in[16]; const float* be2 = (const float*)d_in[17];
  const int*   mask = (const int*)d_in[18];
  float* out = (float*)d_out;

  // Workspace layout (bf16 elements)
  __bf16* ws = (__bf16*)d_ws;
  size_t off = 0;
  __bf16* Wq_bf = ws + off; off += Csz * Csz;
  __bf16* Wk_bf = ws + off; off += Csz * Csz;
  __bf16* Wv_bf = ws + off; off += Csz * Csz;
  __bf16* Wo_bf = ws + off; off += Csz * Csz;
  __bf16* W1_bf = ws + off; off += Fsz * Csz;
  __bf16* W2_bf = ws + off; off += Csz * Fsz;
  __bf16* xln_bf = ws + off; off += (size_t)Msz * Csz;   // reused as attn_bf
  __bf16* qT = ws + off; off += (size_t)Msz * Csz;       // reused as yln_bf
  __bf16* kT = ws + off; off += (size_t)Msz * Csz;
  __bf16* vT = ws + off; off += (size_t)Msz * Csz;
  __bf16* attn_bf = xln_bf;
  __bf16* yln_bf  = qT;

  // 1. weight casts
  cast_f32_bf16_kernel<<<(Csz*Csz + 255) / 256, 256, 0, stream>>>(Wq, Wq_bf, Csz*Csz);
  cast_f32_bf16_kernel<<<(Csz*Csz + 255) / 256, 256, 0, stream>>>(Wk, Wk_bf, Csz*Csz);
  cast_f32_bf16_kernel<<<(Csz*Csz + 255) / 256, 256, 0, stream>>>(Wv, Wv_bf, Csz*Csz);
  cast_f32_bf16_kernel<<<(Csz*Csz + 255) / 256, 256, 0, stream>>>(Wo, Wo_bf, Csz*Csz);
  cast_f32_bf16_kernel<<<(Fsz*Csz + 255) / 256, 256, 0, stream>>>(W1, W1_bf, Fsz*Csz);
  cast_f32_bf16_kernel<<<(Csz*Fsz + 255) / 256, 256, 0, stream>>>(W2, W2_bf, Csz*Fsz);

  // 2. LN1 -> bf16
  ln_cast_kernel<<<Msz, 256, 0, stream>>>(pair_repr, g1, be1, xln_bf);

  // 3. QKV projections (head-transposed outputs)
  qkv_proj_kernel<<<dim3(Msz / 16, 3), 128, 0, stream>>>(
      xln_bf, Wq_bf, Wk_bf, Wv_bf, bq, bk, bv, qT, kT, vT);

  // 4. attention per (j, h)
  attention_kernel<<<dim3(Lsz, Hsz), 128, 0, stream>>>(
      qT, kT, vT, pair_bias, mask, attn_bf);

  // 5. output projection + residual -> pr (fp32 in d_out)
  outproj_kernel<<<Msz / 16, 128, 0, stream>>>(attn_bf, Wo_bf, bo, pair_repr, out);

  // 6. LN2 -> bf16
  ln_cast_kernel<<<Msz, 256, 0, stream>>>(out, g2, be2, yln_bf);

  // 7. fused FFN, out += ff
  ffn_kernel<<<Msz / 16, 128, 0, stream>>>(yln_bf, W1_bf, W2_bf, b1, b2, out);
}
